// ConvPool_67413806678241
// MI455X (gfx1250) — matrix operand order repaired and checked by reading, not compiled, not run on hardware
//
#include <hip/hip_runtime.h>

typedef __attribute__((ext_vector_type(16))) _Float16 v16h;
typedef __attribute__((ext_vector_type(8)))  float    v8f;

#define C_IN  16
#define HW    128
#define OUTC  64
#define NBATCH 32
#define PDIM  63           // pooled spatial dim (126/2)

// x slab layout in LDS: [rr(0..3)][col(0..131)][c(0..15)]  (channel-innermost)
#define XW    132

__global__ __launch_bounds__(256)
void convpool_wmma_kernel(const float* __restrict__ x,
                          const float* __restrict__ w,
                          const float* __restrict__ bias,
                          float* __restrict__ out) {
    __shared__ _Float16 lds_x2[4 * XW * C_IN];     // 8448 * 2B = 16.9 KB
    __shared__ __align__(32) _Float16 lds_pad[16]; // zero pad group (32 B)
    __shared__ _Float16 lds_wB[4 * 5 * 32 * 16];   // 10240 * 2B = 20 KB (B-fragment order)
    __shared__ float    lds_bias[OUTC];

    const int tid = threadIdx.x;
    const int bid = blockIdx.x;
    const int b   = bid / PDIM;       // batch
    const int pr  = bid % PDIM;       // pool row 0..62
    const int r0  = pr * 2;           // first conv row of this block

    // ---- stage bias + zero pad group ----
    if (tid < OUTC) lds_bias[tid] = bias[tid];
    if (tid < 16)   lds_pad[tid]  = (_Float16)0.f;

    // ---- stage x slab (fp32 -> f16), channel-innermost, full width ----
    for (int idx = tid; idx < C_IN * 4 * XW; idx += 256) {
        int c   = idx / (4 * XW);
        int r2  = idx % (4 * XW);
        int rr  = r2 / XW;
        int col = r2 % XW;
        float v = (col < HW) ? x[((b * C_IN + c) * HW + (r0 + rr)) * HW + col] : 0.f;
        lds_x2[(rr * XW + col) * C_IN + c] = (_Float16)v;
    }

    // ---- stage weights pre-packed in per-lane B-fragment order, with the
    //      logical-K permutation matching the contiguous-channel A layout.
    // A slot (half h, vec pos j): tap group g = 2*kc + h, channel c = j
    //   -> L = j*9 + (g/3)*3 + (g%3).
    // B slot (lane', j'): kappa = 16*(lane'>=16) + j' (this code's contiguous k); paired A slot:
//   hA = kappa>>4,  jA = kappa&15.
    for (int idx = tid; idx < 4 * 5 * 32 * 16; idx += 256) {
        int j     = idx & 15;
        int lane  = (idx >> 4) & 31;
        int kc    = (idx >> 9) % 5;
        int nt    = idx / (5 * 32 * 16);
        int kappa = ((lane >= 16) ? 16 : 0) + j;
        int hA    = kappa >> 4;
        int jA    = kappa & 15;
        int g     = 2 * kc + hA;                 // im2col tap group 0..9
        int N     = nt * 16 + (lane & 15);
        float v   = 0.f;
        if (g < 9) v = w[(jA * 9 + (g / 3) * 3 + (g % 3)) * OUTC + N];
        lds_wB[idx] = (_Float16)v;
    }
    __syncthreads();

    const int  wv   = tid >> 5;          // wave 0..7 -> 16-column strip
    const int  lane = tid & 31;
    const bool hi   = lane >= 16;
    const int  lc   = (wv << 4) + (lane & 15);   // this lane's pixel col (A: M = lane%16)

    // ---- build A fragments for BOTH conv rows: contiguous 16-channel v16h each ----
    v16h afr[2][5];
    #pragma unroll
    for (int rs = 0; rs < 2; rs++) {
        #pragma unroll
        for (int kc = 0; kc < 5; kc++) {
            const int glo = 2 * kc;
            const int ghi = 2 * kc + 1;
            const _Float16* plo = &lds_x2[((rs + glo / 3) * XW + lc + glo % 3) * C_IN];
            const _Float16* phi = (ghi < 9)
                ? &lds_x2[((rs + ghi / 3) * XW + lc + ghi % 3) * C_IN]
                : lds_pad;
            const _Float16* p = hi ? phi : plo;
            afr[rs][kc] = *(const v16h*)p;        // 2x ds_load_b128
        }
    }

    // ---- GEMM: 2 row-tiles x 4 N-tiles x 5 K-chunks = 40 WMMAs per wave,
    //      B fragment reused for both rows ----
    v8f acc[2][4];
    #pragma unroll
    for (int rs = 0; rs < 2; rs++)
        #pragma unroll
        for (int nt = 0; nt < 4; nt++)
            #pragma unroll
            for (int r = 0; r < 8; r++) acc[rs][nt][r] = 0.f;

    const v16h* wb = (const v16h*)lds_wB;
    #pragma unroll
    for (int nt = 0; nt < 4; nt++) {
        #pragma unroll
        for (int kc = 0; kc < 5; kc++) {
            v16h bf = wb[(nt * 5 + kc) * 32 + lane];      // 2x ds_load_b128
            acc[0][nt] = __builtin_amdgcn_wmma_f32_16x16x32_f16(
                false, afr[0][kc], false, bf, (short)0, acc[0][nt], false, false);
            acc[1][nt] = __builtin_amdgcn_wmma_f32_16x16x32_f16(
                false, afr[1][kc], false, bf, (short)0, acc[1][nt], false, false);
        }
    }

    // ---- in-register 2x2 max pool + bias + ReLU, straight to global ----
    // lane's D values: pixel col = wv*16 + r + 8*hi; col pair (2q,2q+1) = VGPRs r=2q,2q+1
    const int gpc0 = (wv << 3) + (hi ? 4 : 0);   // first pool col of this lane's group
    #pragma unroll
    for (int nt = 0; nt < 4; nt++) {
        const int   ch    = nt * 16 + (lane & 15);   // C/D: N = lane%16
        const float bv    = lds_bias[ch];
        const int   obase = (b * OUTC + ch) * (PDIM * PDIM) + pr * PDIM + gpc0;
        #pragma unroll
        for (int q = 0; q < 4; q++) {
            float m0 = fmaxf(acc[0][nt][2 * q], acc[0][nt][2 * q + 1]);  // row r0
            float m1 = fmaxf(acc[1][nt][2 * q], acc[1][nt][2 * q + 1]);  // row r0+1
            float v  = fmaxf(fmaxf(m0, m1) + bv, 0.f);   // relu(max) == max(relu)
            if (gpc0 + q < PDIM)                         // drop padding cols 126/127
                out[obase + q] = v;
        }
    }
}

extern "C" void kernel_launch(void* const* d_in, const int* in_sizes, int n_in,
                              void* d_out, int out_size, void* d_ws, size_t ws_size,
                              hipStream_t stream) {
    const float* x    = (const float*)d_in[0];
    const float* w    = (const float*)d_in[1];
    const float* bias = (const float*)d_in[2];
    float*       out  = (float*)d_out;
    dim3 grid(NBATCH * PDIM);   // 2016 blocks: one per (batch, pool row)
    convpool_wmma_kernel<<<grid, 256, 0, stream>>>(x, w, bias, out);
}
